// CrossAttention_55327768707323
// MI455X (gfx1250) — matrix operand order repaired and checked by reading, not compiled, not run on hardware
//
#include <hip/hip_runtime.h>
#include <cstddef>
#include <cstdint>

// ---------------- problem constants (match reference) ----------------
#define B_  2
#define S_  2048
#define H_  2048
#define NH_ 16
#define NKV_ 8
#define HD_ 128
#define EPS_ 1e-6f
#define SCALE_ 0.08838834764831845f   // 128^-0.5

// ---------------- CDNA5 WMMA types ----------------
typedef __attribute__((ext_vector_type(16))) __bf16 v16bf;
typedef __attribute__((ext_vector_type(8)))  __bf16 v8bf;
typedef __attribute__((ext_vector_type(8)))  float  v8f;
typedef __attribute__((ext_vector_type(4)))  float  v4f;

union FragBF { v16bf v; v8bf h[2]; };

// ---------------- CDNA5 async memory->LDS path (ASYNCcnt) ----------------
#if defined(__gfx1250__) && __has_builtin(__builtin_amdgcn_global_load_async_to_lds_b128)
#define USE_ASYNC_LDS 1
#else
#define USE_ASYNC_LDS 0
#endif

#if USE_ASYNC_LDS
// Builtin signature (from hipcc diagnostic): param0 = int4 __device__* (global),
// param1 = LDS pointer, then imm offset + imm cpol.
typedef int v4i_t __attribute__((vector_size(4 * sizeof(int))));
typedef __attribute__((address_space(1))) v4i_t gv4i_t;   // global int4
typedef __attribute__((address_space(3))) v4i_t lv4i_t;   // LDS int4
__device__ __forceinline__ void async_copy_b128(const void* g, void* l) {
  __builtin_amdgcn_global_load_async_to_lds_b128(
      (gv4i_t*)(unsigned long long)g,
      (lv4i_t*)(unsigned int)(unsigned long long)l, 0, 0);
}
template <int N>
__device__ __forceinline__ void wait_async_le() {
#if __has_builtin(__builtin_amdgcn_s_wait_asynccnt)
  __builtin_amdgcn_s_wait_asynccnt(N);
#else
  asm volatile("s_wait_asynccnt %0" :: "i"(N) : "memory");
#endif
}
#endif

// ---------------- WMMA fragment helpers (ISA 7.12.2 layouts) ----------------
// A fragment: 16x32 bf16 tile, row-major source.
__device__ __forceinline__ v16bf load_a_frag(const __bf16* base, int stride, int lane) {
  int m = lane & 15;
  int koff = (lane < 16) ? 0 : 8;
  const __bf16* p = base + m * stride + koff;
  FragBF f;
  f.h[0] = *(const v8bf*)(p);
  f.h[1] = *(const v8bf*)(p + 16);
  return f.v;
}

// B fragment from an N x K row-major source (row n holds contiguous K values).
__device__ __forceinline__ v16bf load_bT_frag(const __bf16* base, int stride, int lane) {
  int n = lane & 15;
  int kbase = (lane < 16) ? 0 : 8;
  const __bf16* p = base + n * stride + kbase;
  FragBF f;
  f.h[0] = *(const v8bf*)(p);
  f.h[1] = *(const v8bf*)(p + 16);
  return f.v;
}

// C/D fragment (16x16 f32), compile-time stride so stores use immediate offsets.
template <int STRIDE>
__device__ __forceinline__ void store_c_frag_f32(float* base, int lane, const v8f& c) {
  int n = lane & 15;
  int mb = (lane < 16) ? 0 : 8;
  float* p = base + mb * STRIDE + n;
#pragma unroll
  for (int r = 0; r < 8; ++r) p[r * STRIDE] = c[r];
}

// ---------------- Kernel 0: vectorized f32 -> bf16 convert ----------------
__global__ __launch_bounds__(256) void cvt_f32_bf16(const float* __restrict__ src,
                                                    __bf16* __restrict__ dst, int n8) {
  int i = blockIdx.x * 256 + threadIdx.x;
  if (i >= n8) return;
  const v4f* s = (const v4f*)src + (size_t)i * 2;
  v4f a = s[0], b = s[1];
  v8bf o;
#pragma unroll
  for (int j = 0; j < 4; ++j) o[j] = (__bf16)a[j];
#pragma unroll
  for (int j = 0; j < 4; ++j) o[4 + j] = (__bf16)b[j];
  ((v8bf*)dst)[i] = o;
}

// ---------------- Kernel 1/4: C[M,N] = A[M,K] * W[N,K]^T, bf16 in, fp32 out ----------
// 128x128 block tile, BK=32, double-buffered LDS with async memory->LDS copies
// pipelined one K-step ahead (partial s_wait_asynccnt).
#define BM 128
#define BN 128
#define BK 32

template <int N, int K>
__global__ __launch_bounds__(256) void gemm_bf16(
    const __bf16* __restrict__ A, const __bf16* __restrict__ W,
    float* __restrict__ C) {
  __shared__ __bf16 As[2][BM][BK];
  __shared__ __bf16 Ws[2][BN][BK];
  const int tid = threadIdx.x;
  const int lane = tid & 31;
  const int wave = tid >> 5;               // 8 waves
  const int m0 = blockIdx.y * BM;
  const int n0 = blockIdx.x * BN;
  const int wr = wave >> 1;                // 0..3 -> rows (wr*2+mi)*16
  const int wc = wave & 1;                 // 0..1 -> cols (wc*4+ni)*16

  v8f acc[2][4] = {};

  // Per-tile staging: A 128x32 = 512 b128 chunks, W 128x32 = 512 chunks (2 each/thread).
  auto issueTile = [&](int kt, int buf) {
#pragma unroll
    for (int c = 0; c < 2; ++c) {
      int idx = tid + c * 256;
      int r = idx >> 2, c0 = (idx & 3) * 8;
#if USE_ASYNC_LDS
      async_copy_b128(A + (size_t)(m0 + r) * K + kt + c0, &As[buf][r][c0]);
      async_copy_b128(W + (size_t)(n0 + r) * K + kt + c0, &Ws[buf][r][c0]);
#else
      *(v8bf*)&As[buf][r][c0] = *(const v8bf*)(A + (size_t)(m0 + r) * K + kt + c0);
      *(v8bf*)&Ws[buf][r][c0] = *(const v8bf*)(W + (size_t)(n0 + r) * K + kt + c0);
#endif
    }
  };

  constexpr int T = K / BK;
  issueTile(0, 0);

  for (int t = 0; t < T; ++t) {
    const int cur = t & 1;
    if (t + 1 < T) issueTile((t + 1) * BK, cur ^ 1);  // prefetch next tile
#if USE_ASYNC_LDS
    if (t + 1 < T) wait_async_le<4>();                // tile t landed (in-order completion)
    else           wait_async_le<0>();
#endif
    __syncthreads();                                  // publish tile t to all waves

    v16bf a0 = load_a_frag(&As[cur][(wr * 2 + 0) * 16][0], BK, lane);
    v16bf a1 = load_a_frag(&As[cur][(wr * 2 + 1) * 16][0], BK, lane);
#pragma unroll
    for (int ni = 0; ni < 4; ++ni) {
      v16bf b = load_bT_frag(&Ws[cur][(wc * 4 + ni) * 16][0], BK, lane);
      acc[0][ni] = __builtin_amdgcn_wmma_f32_16x16x32_bf16(false, a0, false, b, (short)0, acc[0][ni], false, false);
      acc[1][ni] = __builtin_amdgcn_wmma_f32_16x16x32_bf16(false, a1, false, b, (short)0, acc[1][ni], false, false);
    }
    __syncthreads();                                  // all waves done reading buf cur
  }

#pragma unroll
  for (int mi = 0; mi < 2; ++mi)
#pragma unroll
    for (int ni = 0; ni < 4; ++ni)
      store_c_frag_f32<N>(C + (size_t)(m0 + (wr * 2 + mi) * 16) * N + n0 + (wc * 4 + ni) * 16,
                          lane, acc[mi][ni]);
}

// ---------------- Kernel 2: RMS-norm + RoPE (Q,K) and bf16 convert (V) ----------------
__global__ __launch_bounds__(128) void normrope_kernel(
    const float* __restrict__ Qf, const float* __restrict__ Kf, const float* __restrict__ Vf,
    const float* __restrict__ cosb, const float* __restrict__ sinb,
    const float* __restrict__ qw, const float* __restrict__ kw,
    __bf16* __restrict__ Qb, __bf16* __restrict__ Kb, __bf16* __restrict__ Vb) {
  const int row = blockIdx.x;       // b*S + s
  const int slot = blockIdx.y;      // [0,NH): q-head, [NH,NH+NKV): k-head, rest: v convert
  const int d = threadIdx.x;        // 0..127

  if (slot >= NH_ + NKV_) {
    int h = slot - NH_ - NKV_;
    size_t idx = ((size_t)row * NKV_ + h) * HD_ + d;
    Vb[idx] = (__bf16)Vf[idx];
    return;
  }

  __shared__ float xs[HD_];
  __shared__ float red[4];

  const bool isq = slot < NH_;
  const int h = isq ? slot : slot - NH_;
  const float* X = isq ? Qf : Kf;
  const float* w = isq ? qw : kw;
  __bf16* Y = isq ? Qb : Kb;
  const int nh = isq ? NH_ : NKV_;
  const size_t base = ((size_t)row * nh + h) * HD_;

  float x = X[base + d];
  float ss = x * x;
#pragma unroll
  for (int off = 16; off > 0; off >>= 1) ss += __shfl_xor(ss, off, 32);
  if ((d & 31) == 0) red[d >> 5] = ss;
  __syncthreads();
  float tot = red[0] + red[1] + red[2] + red[3];
  float rs = rsqrtf(tot * (1.0f / HD_) + EPS_);
  float xn = x * rs * w[d];
  xs[d] = xn;
  __syncthreads();

  float c = cosb[(size_t)row * HD_ + d];
  float s = sinb[(size_t)row * HD_ + d];
  float rot = (d < HD_ / 2) ? -xs[d + HD_ / 2] : xs[d - HD_ / 2];
  Y[base + d] = (__bf16)(xn * c + rot * s);
}

// ---------------- Kernel 3: attention (scores + exact softmax + PV) ----------------
// One workgroup per (b, h, 16-row q tile). Full 2048-wide score rows in LDS
// (204 KB dynamic LDS, possible thanks to CDNA5's 320 KB/WGP LDS).
__global__ __launch_bounds__(256) void attn_kernel(
    const __bf16* __restrict__ Qb, const __bf16* __restrict__ Kb,
    const __bf16* __restrict__ Vb, const unsigned char* __restrict__ mask,
    float* __restrict__ attnw, __bf16* __restrict__ aout) {
  extern __shared__ char smem[];
  float*  Ssc = (float*)smem;                                        // [16][S_] fp32
  __bf16* P   = (__bf16*)(smem + (size_t)16 * S_ * 4);               // [16][S_] bf16
  __bf16* Qt  = (__bf16*)(smem + (size_t)16 * S_ * 4 + 16 * S_ * 2); // [16][HD_]
  __bf16* VtT = (__bf16*)(smem + (size_t)16 * S_ * 4 + 16 * S_ * 2 + 16 * HD_ * 2); // [HD_][32]

  const int q0 = blockIdx.x * 16;
  const int h = blockIdx.y;
  const int b = blockIdx.z;
  const int hk = h / (NH_ / NKV_);
  const int tid = threadIdx.x;
  const int lane = tid & 31;
  const int wave = tid >> 5;

  // Stage the 16x128 q tile (async memory->LDS when available).
  {
    int r = tid >> 4, c0 = (tid & 15) * 8;
    const __bf16* src = Qb + ((size_t)(b * S_ + q0 + r) * NH_ + h) * HD_ + c0;
#if USE_ASYNC_LDS
    async_copy_b128(src, &Qt[r * HD_ + c0]);
    wait_async_le<0>();
#else
    *(v8bf*)&Qt[r * HD_ + c0] = *(const v8bf*)src;
#endif
  }
  __syncthreads();

  // ---- Phase 1: scores = scale * Q K^T, into LDS. A-fragments hoisted. ----
  const __bf16* Kbase = Kb + (size_t)b * S_ * NKV_ * HD_ + hk * HD_; // row s stride NKV_*HD_
  v16bf aq[4];
#pragma unroll
  for (int dc = 0; dc < 4; ++dc) aq[dc] = load_a_frag(Qt + dc * 32, HD_, lane);

  for (int i = 0; i < 16; ++i) {
    int ct = wave * 16 + i;                     // column tile 0..127
    v8f acc = {};
    const __bf16* kb = Kbase + (size_t)(ct * 16) * (NKV_ * HD_);
#pragma unroll
    for (int dc = 0; dc < 4; ++dc) {
      v16bf bf = load_bT_frag(kb + dc * 32, NKV_ * HD_, lane);
      acc = __builtin_amdgcn_wmma_f32_16x16x32_bf16(false, aq[dc], false, bf, (short)0, acc,
                                                    false, false);
    }
    int n = lane & 15, mb = (lane < 16) ? 0 : 8;
#pragma unroll
    for (int r = 0; r < 8; ++r)
      Ssc[(mb + r) * S_ + ct * 16 + n] = acc[r] * SCALE_;
  }
  __syncthreads();

  // ---- Phase 2: exact fp32 softmax per row (one wave per 2 rows) ----
  const unsigned char* mrow = mask + (size_t)b * S_;
  for (int rr = 0; rr < 2; ++rr) {
    int r = wave * 2 + rr;
    float mx = -3.402823466e38f;
    for (int c = lane; c < S_; c += 32) {
      float v = Ssc[r * S_ + c];
      v += mrow[c] ? 0.0f : -3.402823466e38f;
      Ssc[r * S_ + c] = v;
      mx = fmaxf(mx, v);
    }
#pragma unroll
    for (int off = 16; off > 0; off >>= 1) mx = fmaxf(mx, __shfl_xor(mx, off, 32));
    float sum = 0.0f;
    for (int c = lane; c < S_; c += 32) {
      float e = __expf(Ssc[r * S_ + c] - mx);
      Ssc[r * S_ + c] = e;
      sum += e;
    }
#pragma unroll
    for (int off = 16; off > 0; off >>= 1) sum += __shfl_xor(sum, off, 32);
    float inv = 1.0f / sum;
    float* wout = attnw + ((size_t)(b * NH_ + h) * S_ + q0 + r) * S_;
    for (int c = lane; c < S_; c += 32) {
      float p = Ssc[r * S_ + c] * inv;
      wout[c] = p;                 // fp32 attn_weights output
      P[r * S_ + c] = (__bf16)p;   // bf16 copy for PV matmul
    }
  }
  __syncthreads();

  // ---- Phase 3: attn_out(16x128) = P(16xS) * V(SxHD_); wave w owns d-cols [16w,16w+16).
  // V chunks are staged TRANSPOSED so B-fragments are two contiguous ds_load_b128.
  v8f acc = {};
  const __bf16* Vbase = Vb + (size_t)b * S_ * NKV_ * HD_ + hk * HD_;
  for (int kt = 0; kt < S_; kt += 32) {
    __syncthreads();
    {
      int r = tid >> 3;            // 0..31 source row within chunk
      int c0 = (tid & 7) * 16;     // 0..112
      const __bf16* src = Vbase + (size_t)(kt + r) * (NKV_ * HD_) + c0;
      if (kt + 32 < S_) __builtin_prefetch(src + 32 * (NKV_ * HD_), 0, 0);
      v8bf x0 = *(const v8bf*)src;
      v8bf x1 = *(const v8bf*)(src + 8);
#pragma unroll
      for (int j = 0; j < 8; ++j) VtT[(c0 + j) * 32 + r] = x0[j];
#pragma unroll
      for (int j = 0; j < 8; ++j) VtT[(c0 + 8 + j) * 32 + r] = x1[j];
    }
    __syncthreads();
    v16bf a  = load_a_frag(P + kt, S_, lane);
    v16bf bf = load_bT_frag(VtT + (wave * 16) * 32, 32, lane);
    acc = __builtin_amdgcn_wmma_f32_16x16x32_bf16(false, a, false, bf, (short)0, acc,
                                                  false, false);
  }
  {
    int n = lane & 15, mb = (lane < 16) ? 0 : 8;
    __bf16* dst = aout + (size_t)(b * S_ + q0 + mb) * (NH_ * HD_) + h * HD_ + wave * 16 + n;
#pragma unroll
    for (int r = 0; r < 8; ++r) dst[r * (NH_ * HD_)] = (__bf16)acc[r];
  }
}

// ---------------- host-side orchestration ----------------
extern "C" void kernel_launch(void* const* d_in, const int* in_sizes, int n_in,
                              void* d_out, int out_size, void* d_ws, size_t ws_size,
                              hipStream_t stream) {
  (void)in_sizes; (void)n_in; (void)out_size; (void)ws_size;

  const float* hidden = (const float*)d_in[0];
  const float* mm     = (const float*)d_in[1];
  const float* cosb   = (const float*)d_in[2];
  const float* sinb   = (const float*)d_in[3];
  const unsigned char* mask = (const unsigned char*)d_in[4];
  const float* Wq = (const float*)d_in[5];
  const float* Wk = (const float*)d_in[6];
  const float* Wv = (const float*)d_in[7];
  const float* Wo = (const float*)d_in[8];
  const float* qw = (const float*)d_in[9];
  const float* kw = (const float*)d_in[10];

  // workspace layout (bytes)
  char* ws = (char*)d_ws;
  __bf16* Xh_b = (__bf16*)(ws);                 // 4096x2048 bf16  (16 MB)
  __bf16* Xm_b = (__bf16*)(ws + 16777216);      // 4096x2048 bf16
  __bf16* Wq_b = (__bf16*)(ws + 33554432);      // 2048x2048 bf16  (8 MB)
  __bf16* Wk_b = (__bf16*)(ws + 41943040);      // 1024x2048 bf16  (4 MB)
  __bf16* Wv_b = (__bf16*)(ws + 46137344);      // 1024x2048 bf16
  __bf16* Wo_b = (__bf16*)(ws + 50331648);      // 2048x2048 bf16
  float*  Qf   = (float*)(ws + 58720256);       // 4096x2048 fp32  (32 MB)
  float*  Kf   = (float*)(ws + 92274688);       // 4096x1024 fp32  (16 MB)
  float*  Vf   = (float*)(ws + 109051904);      // 4096x1024 fp32
  __bf16* Qb   = (__bf16*)(ws + 125829120);     // 4096x2048 bf16 (post norm+rope)
  __bf16* Kb   = (__bf16*)(ws + 142606336);     // 4096x1024 bf16
  __bf16* Vb   = (__bf16*)(ws + 150994944);     // 4096x1024 bf16
  __bf16* Ab   = (__bf16*)Qf;                   // attn_out bf16 reuses dead Qf region
  // total ws used: 159,383,552 bytes

  float* outp  = (float*)d_out;
  float* attnw = outp + (size_t)B_ * S_ * H_;

  const int M = B_ * S_;                        // 4096
  dim3 blk(256);

  // bf16 pre-conversion (all operands are L2-resident; one cheap pass each)
  cvt_f32_bf16<<<dim3(4096), blk, 0, stream>>>(hidden, Xh_b, (B_ * S_ * H_) / 8);
  cvt_f32_bf16<<<dim3(4096), blk, 0, stream>>>(mm, Xm_b, (B_ * S_ * H_) / 8);
  cvt_f32_bf16<<<dim3(2048), blk, 0, stream>>>(Wq, Wq_b, (NH_ * HD_ * H_) / 8);
  cvt_f32_bf16<<<dim3(1024), blk, 0, stream>>>(Wk, Wk_b, (NKV_ * HD_ * H_) / 8);
  cvt_f32_bf16<<<dim3(1024), blk, 0, stream>>>(Wv, Wv_b, (NKV_ * HD_ * H_) / 8);
  cvt_f32_bf16<<<dim3(2048), blk, 0, stream>>>(Wo, Wo_b, (H_ * NH_ * HD_) / 8);

  // QKV projections (bf16 WMMA, fp32 accumulate)
  gemm_bf16<NH_ * HD_, H_><<<dim3((NH_ * HD_) / BN, M / BM), blk, 0, stream>>>(Xh_b, Wq_b, Qf);
  gemm_bf16<NKV_ * HD_, H_><<<dim3((NKV_ * HD_) / BN, M / BM), blk, 0, stream>>>(Xm_b, Wk_b, Kf);
  gemm_bf16<NKV_ * HD_, H_><<<dim3((NKV_ * HD_) / BN, M / BM), blk, 0, stream>>>(Xm_b, Wv_b, Vf);

  // RMS norm + RoPE + bf16 convert
  normrope_kernel<<<dim3(M, NH_ + 2 * NKV_), dim3(128), 0, stream>>>(
      Qf, Kf, Vf, cosb, sinb, qw, kw, Qb, Kb, Vb);

  // attention: 208,896-byte dynamic LDS
  size_t smemBytes = (size_t)16 * S_ * 4 + (size_t)16 * S_ * 2 + 16 * HD_ * 2 + HD_ * 32 * 2;
  attn_kernel<<<dim3(S_ / 16, NH_, B_), blk, smemBytes, stream>>>(
      Qb, Kb, Vb, mask, attnw, Ab);

  // output projection: out = attn_out @ Wo^T
  gemm_bf16<H_, H_><<<dim3(H_ / BN, M / BM), blk, 0, stream>>>(Ab, Wo_b, outp);
}
